// GcnEdgeAngleConv2_39840116637828
// MI455X (gfx1250) — matrix operand order repaired and checked by reading, not compiled, not run on hardware
//
#include <hip/hip_runtime.h>
#include <math.h>

typedef __attribute__((ext_vector_type(16))) __bf16 v16bf;
typedef __attribute__((ext_vector_type(8)))  float  v8f;

#define NNODE 2048
#define NEDGE 4096
#define TWOE  8192
#define HW    64
#define F1D   10

// ---------------------------------------------------------------------------
// Prep: directed edge lists, gate = w*cos(angle)
// ---------------------------------------------------------------------------
__global__ void prep_edges(const int* __restrict__ ei, const float* __restrict__ ew,
                           const float* __restrict__ ang,
                           int* __restrict__ src2, int* __restrict__ dst2,
                           float* __restrict__ gate) {
  int i = blockIdx.x * blockDim.x + threadIdx.x;
  if (i >= TWOE) return;
  int s, d, e;
  if (i < NEDGE) { e = i;         s = ei[e];         d = ei[NEDGE + e]; }
  else           { e = i - NEDGE; s = ei[NEDGE + e]; d = ei[e];         }
  src2[i] = s; dst2[i] = d;
  gate[i] = ew[e] * cosf(ang[i]);
}

// ---------------------------------------------------------------------------
// Pack conv weights [Cout][Cin0+Cin1][3][3] f32 -> bf16 [Cout][Ctot*16]
// (9 taps padded to 16). symHalf>0: Wsym = W + swapChannelHalves(W)
// (folds the two edge directions into one conv).
// ---------------------------------------------------------------------------
__global__ void pack_weights(const float* __restrict__ W0, const float* __restrict__ W1,
                             __bf16* __restrict__ Wp, int Cout, int Cin0_, int Cin1_,
                             int symHalf) {
  int Ctot = Cin0_ + Cin1_;
  long total = (long)Cout * Ctot * 16;
  long i = (long)blockIdx.x * blockDim.x + threadIdx.x;
  if (i >= total) return;
  int t = (int)(i & 15);
  long ci = i >> 4;
  int c = (int)(ci % Ctot);
  int o = (int)(ci / Ctot);
  float v = 0.0f;
  if (t < 9) {
    if (c < Cin0_) {
      v = W0[((long)o * Cin0_ + c) * 9 + t];
      if (symHalf > 0) {
        int cs = (c < symHalf) ? c + symHalf : c - symHalf;
        v += W0[((long)o * Cin0_ + cs) * 9 + t];
      }
    } else {
      v = W1[((long)o * Cin1_ + (c - Cin0_)) * 9 + t];
    }
  }
  Wp[i] = (__bf16)v;
}

__global__ void add_vec(const float* __restrict__ a, const float* __restrict__ b,
                        float* __restrict__ o, int n) {
  int i = blockIdx.x * blockDim.x + threadIdx.x;
  if (i < n) o[i] = a[i] + b[i];
}

// Collapse the two stacked linears (no activation between): Wc = Wl2 @ Wl1 [2][139]
__global__ void head_precompute(const float* __restrict__ Wl1, const float* __restrict__ bl1,
                                const float* __restrict__ Wl2, const float* __restrict__ bl2,
                                float* __restrict__ Wc, float* __restrict__ bc) {
  int t = threadIdx.x;
  if (t < 2 * 139) {
    int j = t / 139, k = t % 139;
    float s = 0.0f;
    for (int i = 0; i < 256; ++i) s += Wl2[j * 256 + i] * Wl1[i * 139 + k];
    Wc[j * 139 + k] = s;
  } else if (t < 2 * 139 + 2) {
    int j = t - 2 * 139;
    float s = bl2[j];
    for (int i = 0; i < 256; ++i) s += Wl2[j * 256 + i] * bl1[i];
    bc[j] = s;
  }
}

// ---------------------------------------------------------------------------
// agg[dst[e]] += X[src[e]] * gate[e]
// ---------------------------------------------------------------------------
__global__ void scatter_agg(const float* __restrict__ X, const int* __restrict__ src,
                            const int* __restrict__ dst, const float* __restrict__ gate,
                            float* __restrict__ agg, int C) {
  long i = (long)blockIdx.x * blockDim.x + threadIdx.x;
  long tot = (long)TWOE * C * HW;
  if (i >= tot) return;
  int cw = C * HW;
  int e = (int)(i / cw);
  int r = (int)(i % cw);
  atomicAdd(&agg[(size_t)dst[e] * cw + r], X[(size_t)src[e] * cw + r] * gate[e]);
}

// ---------------------------------------------------------------------------
// Pipelined K-loop: double-buffered B fragments (global) and A fragments
// (LDS) so the v_wmma bursts never wait on fresh loads.
// Requires Ctot % 4 == 0 (all our convs: 16 / 128 / 256).
// ---------------------------------------------------------------------------
template<int NCT>
__device__ __forceinline__
void conv_k_loop(const __bf16* __restrict__ ldsbuf, int Ctot,
                 const __bf16* __restrict__ wbase, int Kpad,
                 const int (&aoff)[8], v8f (&acc)[NCT]) {
  auto loadB = [&](v16bf (&buf)[NCT], int c0) {
    const __bf16* wb = wbase + (size_t)c0 * 16;
#pragma unroll
    for (int ct = 0; ct < NCT; ++ct) {
      const __bf16* pb = wb + (size_t)ct * 16 * Kpad;
#pragma unroll
      for (int s = 0; s < 8; ++s) { buf[ct][s] = pb[s]; buf[ct][s + 8] = pb[16 + s]; }
    }
  };
  auto loadA = [&](int c0) -> v16bf {
    const __bf16* ch0 = ldsbuf + c0 * 100;
    v16bf a;
#pragma unroll
    for (int t = 0; t < 8; ++t) { a[t] = ch0[aoff[t]]; a[t + 8] = ch0[100 + aoff[t]]; }
    return a;
  };

  v16bf b0[NCT], b1[NCT];
  loadB(b0, 0);
  v16bf a0 = loadA(0);
  for (int c0 = 0; c0 < Ctot; c0 += 4) {
    // prefetch step c0+2 while computing step c0
    loadB(b1, c0 + 2);
    v16bf a1 = loadA(c0 + 2);
#pragma unroll
    for (int ct = 0; ct < NCT; ++ct)
      acc[ct] = __builtin_amdgcn_wmma_f32_16x16x32_bf16(false, a0, false, b0[ct],
                                                        (short)0, acc[ct], false, false);
    // prefetch step c0+4 (clamped; extra loads of step 0 on last iter are harmless)
    int nx = (c0 + 4 < Ctot) ? (c0 + 4) : 0;
    loadB(b0, nx);
    a0 = loadA(nx);
#pragma unroll
    for (int ct = 0; ct < NCT; ++ct)
      acc[ct] = __builtin_amdgcn_wmma_f32_16x16x32_bf16(false, a1, false, b1[ct],
                                                        (short)0, acc[ct], false, false);
  }
}

// ---------------------------------------------------------------------------
// Implicit-GEMM 3x3 SAME conv on 8x8 images, v_wmma_f32_16x16x32_bf16.
// One block = one image, ALL Cout = NCT*16 channels. 4 waves = 4 pixel tiles.
// Input = [X0[idx0] (Cin0) ; X1[idx1] (Cin1)] staged in LDS as bf16 with a
// zero halo [ch][10*10]; branchless A-fragment (invalid taps read a zero cell).
// HASB: acc=(acc+biasA)*scale, restage LDS with X2[img], accumulate a second
// conv (WpB) on top (fused residual). POOL: fused mean-pool epilogue.
// ---------------------------------------------------------------------------
template<int NCT, bool HASB, bool POOL>
__global__ __launch_bounds__(128)
void conv3x3_wmma(const float* __restrict__ X0, const float* __restrict__ X1,
                  const float* __restrict__ X2,
                  const int* __restrict__ idx0, const int* __restrict__ idx1,
                  const __bf16* __restrict__ WpA, const __bf16* __restrict__ WpB,
                  const float* __restrict__ biasA, const float* __restrict__ biasB,
                  const float* __restrict__ scale, float* __restrict__ Y,
                  int Cin0, int Cin1, int Cin2, int doRelu) {
  extern __shared__ __bf16 lds[];            // [max(CtotA,Cin2)][100]
  const int img = blockIdx.x;
  const int tid = threadIdx.x;
  const int CtotA = Cin0 + Cin1;

  for (int i = tid; i < CtotA * 100; i += 128) lds[i] = (__bf16)0.0f;
  __syncthreads();
  const float* s0 = X0 + (size_t)(idx0 ? idx0[img] : img) * Cin0 * HW;
  const float* s1 = Cin1 ? (X1 + (size_t)(idx1 ? idx1[img] : img) * Cin1 * HW)
                         : (const float*)0;
  for (int i = tid; i < CtotA * HW; i += 128) {
    int c = i >> 6, p = i & 63;
    float v = (c < Cin0) ? s0[c * HW + p] : s1[(c - Cin0) * HW + p];
    lds[c * 100 + ((p >> 3) + 1) * 10 + (p & 7) + 1] = (__bf16)v;
  }
  __syncthreads();

  const int lane = tid & 31;
  const int l    = lane & 15;
  const int hi   = lane >> 4;
  const int mrow = (tid >> 5) * 16;
  const int p    = mrow + l;
  const int abase = ((p >> 3) + 1) * 10 + (p & 7) + 1;

  // Branchless per-lane A offsets: hi half has only tap 8 valid; the rest
  // read lds[ch*100 + 0], a zeroed halo cell.
  int aoff[8];
#pragma unroll
  for (int t = 0; t < 8; ++t) {
    int o0 = abase + (t / 3) * 10 + (t % 3) - 11;
    aoff[t] = hi ? ((t == 0) ? (abase + 11) : 0) : o0;
  }

  v8f zero = {};
  v8f acc[NCT];
#pragma unroll
  for (int ct = 0; ct < NCT; ++ct) acc[ct] = zero;

  const int KpadA = CtotA * 16;
  const __bf16* wbaseA = WpA + (size_t)l * KpadA + (size_t)hi * 8;
  conv_k_loop<NCT>(lds, CtotA, wbaseA, KpadA, aoff, acc);

  const float sc = scale ? scale[img] : 1.0f;

  if constexpr (HASB) {
    // finish part A: acc = (acc + biasA) * scale, then fuse residual conv
#pragma unroll
    for (int ct = 0; ct < NCT; ++ct) {
      float bA = biasA ? biasA[ct * 16 + l] : 0.0f;
#pragma unroll
      for (int i = 0; i < 8; ++i) acc[ct][i] = (acc[ct][i] + bA) * sc;
    }
    __syncthreads();                       // all waves done reading part-A LDS
    const float* s2 = X2 + (size_t)img * Cin2 * HW;
    for (int i = tid; i < Cin2 * HW; i += 128) {
      int c = i >> 6, pp = i & 63;
      lds[c * 100 + ((pp >> 3) + 1) * 10 + (pp & 7) + 1] = (__bf16)s2[i];
    }
    __syncthreads();
    const int KpadB = Cin2 * 16;
    const __bf16* wbaseB = WpB + (size_t)l * KpadB + (size_t)hi * 8;
    conv_k_loop<NCT>(lds, Cin2, wbaseB, KpadB, aoff, acc);
  }

#pragma unroll
  for (int ct = 0; ct < NCT; ++ct) {
    const int och = ct * 16 + l;
    float v[8];
#pragma unroll
    for (int i = 0; i < 8; ++i) {
      float x;
      if constexpr (HASB) x = acc[ct][i] + (biasB ? biasB[och] : 0.0f);
      else                x = (acc[ct][i] + (biasA ? biasA[och] : 0.0f)) * sc;
      v[i] = doRelu ? fmaxf(x, 0.0f) : x;
    }
    if constexpr (POOL) {
      float ps = ((v[0] + v[1]) + (v[2] + v[3])) + ((v[4] + v[5]) + (v[6] + v[7]));
      ps += __shfl_xor(ps, 16, 32);        // combine the two pixel half-tiles
      if (hi == 0)
        atomicAdd(&Y[(size_t)img * (NCT * 16) + och], ps * (1.0f / 64.0f));
    } else {
      float* yo = Y + ((size_t)img * (NCT * 16) + och) * HW + mrow + hi * 8;
      float4 f0 = {v[0], v[1], v[2], v[3]};
      float4 f1 = {v[4], v[5], v[6], v[7]};
      *(float4*)yo = f0;
      *(float4*)(yo + 4) = f1;
    }
  }
}

// ---------------------------------------------------------------------------
__global__ void head_kernel(const float* __restrict__ pooled, const float* __restrict__ ef1d,
                            const float* __restrict__ ew, const float* __restrict__ Wc,
                            const float* __restrict__ bc, float* __restrict__ out) {
  int e = blockIdx.x * blockDim.x + threadIdx.x;
  if (e >= NEDGE) return;
  float l0 = bc[0], l1 = bc[1];
  const float* pr = pooled + (long)e * 128;
  for (int k = 0; k < 128; ++k) { float v = pr[k]; l0 += Wc[k] * v; l1 += Wc[139 + k] * v; }
  const float* fr = ef1d + (long)e * F1D;
  for (int k = 0; k < F1D; ++k) { float v = fr[k]; l0 += Wc[128 + k] * v; l1 += Wc[139 + 128 + k] * v; }
  float wv = ew[e];
  l0 += Wc[138] * wv; l1 += Wc[139 + 138] * wv;
  float m = fmaxf(l0, l1);
  float e0 = expf(l0 - m), e1 = expf(l1 - m);
  float inv = 1.0f / (e0 + e1);
  out[2 * e]     = e0 * inv;
  out[2 * e + 1] = e1 * inv;
}

// ---------------------------------------------------------------------------
extern "C" void kernel_launch(void* const* d_in, const int* in_sizes, int n_in,
                              void* d_out, int out_size, void* d_ws, size_t ws_size,
                              hipStream_t stream) {
  const float* node_features = (const float*)d_in[0];
  const float* ef1d     = (const float*)d_in[1];
  const int*   eidx     = (const int*)d_in[2];
  const float* angles   = (const float*)d_in[3];
  const float* eweights = (const float*)d_in[4];
  const float* Wself1   = (const float*)d_in[5];
  const float* bself1   = (const float*)d_in[6];
  const float* Wnbr1    = (const float*)d_in[7];
  const float* bnbr1    = (const float*)d_in[8];
  const float* We1      = (const float*)d_in[9];
  const float* be1      = (const float*)d_in[10];
  const float* Wself2   = (const float*)d_in[11];
  const float* bself2   = (const float*)d_in[12];
  const float* Wnbr2    = (const float*)d_in[13];
  const float* bnbr2    = (const float*)d_in[14];
  const float* We2      = (const float*)d_in[15];
  const float* be2      = (const float*)d_in[16];
  const float* Winit    = (const float*)d_in[17];
  const float* binit    = (const float*)d_in[18];
  const float* Wl1      = (const float*)d_in[19];
  const float* bl1      = (const float*)d_in[20];
  const float* Wl2      = (const float*)d_in[21];
  const float* bl2      = (const float*)d_in[22];

  char* base = (char*)d_ws;
  size_t off = 0;
  auto alloc = [&](size_t bytes) -> char* {
    char* ptr = base + off;
    off = (off + bytes + 255) & ~(size_t)255;
    return ptr;
  };
  int*    src2   = (int*)   alloc((size_t)TWOE * 4);
  int*    dst2   = (int*)   alloc((size_t)TWOE * 4);
  float*  gate   = (float*) alloc((size_t)TWOE * 4);
  float*  bsum1  = (float*) alloc(64 * 4);
  float*  bsum2  = (float*) alloc(128 * 4);
  float*  be1x2  = (float*) alloc(128 * 4);
  float*  be2x2  = (float*) alloc(128 * 4);
  float*  Wc     = (float*) alloc(2 * 139 * 4);
  float*  bc     = (float*) alloc(2 * 4);
  __bf16* Wp1    = (__bf16*)alloc((size_t)64 * 256 * 2);      // node1: Ctot=16
  __bf16* Wpe1   = (__bf16*)alloc((size_t)128 * 2048 * 2);    // edge1 sym: Ctot=128
  __bf16* Wp2    = (__bf16*)alloc((size_t)128 * 2048 * 2);    // node2: Ctot=128
  __bf16* Wpe2   = (__bf16*)alloc((size_t)128 * 4096 * 2);    // edge2 sym: Ctot=256
  __bf16* Wpinit = (__bf16*)alloc((size_t)128 * 2048 * 2);    // init: Ctot=128
  float*  agg1   = (float*) alloc((size_t)NNODE * 8   * HW * 4);
  float*  x1     = (float*) alloc((size_t)NNODE * 64  * HW * 4);
  float*  agg2   = (float*) alloc((size_t)NNODE * 64  * HW * 4);
  float*  x2     = (float*) alloc((size_t)NNODE * 128 * HW * 4);
  float*  ef1    = (float*) alloc((size_t)NEDGE * 128 * HW * 4);
  float*  pooled = (float*) alloc((size_t)NEDGE * 128 * 4);

  // ---- prep ----
  prep_edges<<<(TWOE + 255) / 256, 256, 0, stream>>>(eidx, eweights, angles,
                                                     src2, dst2, gate);
  pack_weights<<<(64 * 16 * 16 + 255) / 256, 256, 0, stream>>>(Wself1, Wnbr1, Wp1, 64, 8, 8, 0);
  pack_weights<<<(128 * 128 * 16 + 255) / 256, 256, 0, stream>>>(We1, nullptr, Wpe1, 128, 128, 0, 64);
  pack_weights<<<(128 * 128 * 16 + 255) / 256, 256, 0, stream>>>(Wself2, Wnbr2, Wp2, 128, 64, 64, 0);
  pack_weights<<<(128 * 256 * 16 + 255) / 256, 256, 0, stream>>>(We2, nullptr, Wpe2, 128, 256, 0, 128);
  pack_weights<<<(128 * 128 * 16 + 255) / 256, 256, 0, stream>>>(Winit, nullptr, Wpinit, 128, 128, 0, 0);
  add_vec<<<1, 64, 0, stream>>>(bself1, bnbr1, bsum1, 64);
  add_vec<<<1, 128, 0, stream>>>(bself2, bnbr2, bsum2, 128);
  add_vec<<<1, 128, 0, stream>>>(be1, be1, be1x2, 128);
  add_vec<<<1, 128, 0, stream>>>(be2, be2, be2x2, 128);
  head_precompute<<<1, 320, 0, stream>>>(Wl1, bl1, Wl2, bl2, Wc, bc);

  hipMemsetAsync(agg1, 0, (size_t)NNODE * 8 * HW * 4, stream);
  hipMemsetAsync(agg2, 0, (size_t)NNODE * 64 * HW * 4, stream);
  hipMemsetAsync(pooled, 0, (size_t)NEDGE * 128 * 4, stream);

  // ---- layer 1: node conv (8 -> 64) ----
  {
    long tot = (long)TWOE * 8 * HW;
    scatter_agg<<<(int)((tot + 255) / 256), 256, 0, stream>>>(node_features, src2, dst2,
                                                              gate, agg1, 8);
  }
  conv3x3_wmma<4, false, false><<<NNODE, 128, 16 * 200, stream>>>(
      node_features, agg1, nullptr, nullptr, nullptr, Wp1, nullptr,
      bsum1, nullptr, nullptr, x1, 8, 8, 0, 1);

  // ---- layer 1: edge conv, both directions folded via symmetric weights ----
  conv3x3_wmma<8, false, false><<<NEDGE, 128, 128 * 200, stream>>>(
      x1, x1, nullptr, eidx, eidx + NEDGE, Wpe1, nullptr,
      be1x2, nullptr, eweights, ef1, 64, 64, 0, 1);

  // ---- layer 2: node conv (64 -> 128) ----
  {
    long tot = (long)TWOE * 64 * HW;
    scatter_agg<<<(int)((tot + 255) / 256), 256, 0, stream>>>(x1, src2, dst2, gate, agg2, 64);
  }
  conv3x3_wmma<8, false, false><<<NNODE, 128, 128 * 200, stream>>>(
      x1, agg2, nullptr, nullptr, nullptr, Wp2, nullptr,
      bsum2, nullptr, nullptr, x2, 64, 64, 0, 1);

  // ---- layer 2: edge conv (sym-folded) + fused init conv + fused mean-pool ----
  conv3x3_wmma<8, true, true><<<NEDGE, 128, 256 * 200, stream>>>(
      x2, x2, ef1, eidx, eidx + NEDGE, Wpe2, Wpinit,
      be2x2, binit, eweights, pooled, 128, 128, 128, 1);

  // ---- head ----
  head_kernel<<<(NEDGE + 255) / 256, 256, 0, stream>>>(pooled, ef1d, eweights, Wc, bc,
                                                       (float*)d_out);
}
